// FXT_56564719288851
// MI455X (gfx1250) — compile-verified, matrix operand order checked by reading, not run on hardware
//
#include <hip/hip_runtime.h>
#include <hip/hip_bf16.h>
#include <math.h>

typedef _Float16 f16;
typedef __attribute__((ext_vector_type(16))) _Float16 v16h;
typedef __attribute__((ext_vector_type(8)))  float    v8f;

// Model constants
#define BB   16
#define SS   511
#define FFT  128
#define DD   512
#define HH   8
#define DHD  64
#define NBLK 6
#define MLPD 2048
#define TT   512

union U4H8 { uint4 u; f16 h[8]; };
union U2H4 { uint2 u; f16 h[4]; };
union H2U  { uint  u; f16 h[2]; };

// ISA 7.12.2: for fixed lane-half, a 16-bit fragment is two contiguous
// 8xf16 chunks: K = 8*half + [0..7]  and  K = 16 + 8*half + [0..7].
__device__ __forceinline__ v16h frag_from_chunks(const f16* p0, const f16* p1) {
  U4H8 lo, hi;
  lo.u = *(const uint4*)p0;
  hi.u = *(const uint4*)p1;
  v16h f;
  #pragma unroll
  for (int e = 0; e < 8; ++e) { f[e] = lo.h[e]; f[8 + e] = hi.h[e]; }
  return f;
}

// ---------------------------------------------------------------------------
// f32 -> f16 convert (grid-stride)
// ---------------------------------------------------------------------------
__global__ void f32_to_f16(const float* __restrict__ in, f16* __restrict__ out, long n) {
  long i = (long)blockIdx.x * blockDim.x + threadIdx.x;
  long stride = (long)gridDim.x * blockDim.x;
  for (; i < n; i += stride) out[i] = (f16)in[i];
}

// ---------------------------------------------------------------------------
// Generic WMMA GEMM: C[M,N] = act(A[M,K] * B[K,N] + bias)
// Block tile 256x64 (8 waves); each wave: 32 rows x 64 cols = 8 WMMA tiles,
// K-step 32. A/B staged in LDS; B transposed [col][k] so fragments are two
// contiguous b128 chunks. ACT: 0=none, 1=gelu(exact). ADDC: C32 += result.
// OUT16: write f16. blockIdx.z applies per-z offsets (per-head QKV).
// ---------------------------------------------------------------------------
template<int ACT, bool ADDC, bool OUT16>
__global__ __launch_bounds__(256)
void wmma_gemm(const f16* __restrict__ A, int lda,
               const f16* __restrict__ Bm, int ldb,
               float* __restrict__ C32, f16* __restrict__ C16, int ldc,
               const float* __restrict__ bias,
               int M, int N, int K,
               int aZ, int bZ, int cZ, int biasZ)
{
  A  += (size_t)blockIdx.z * aZ;
  Bm += (size_t)blockIdx.z * bZ;
  if (C32)  C32  += (size_t)blockIdx.z * cZ;
  if (C16)  C16  += (size_t)blockIdx.z * cZ;
  if (bias) bias += (size_t)blockIdx.z * biasZ;

  __shared__ __align__(16) f16 As[256 * 32];
  __shared__ __align__(16) f16 Bs[64 * 32];   // transposed: [col][k]

  const int tid  = threadIdx.x;
  const int wv   = tid >> 5;
  const int lane = tid & 31;
  const int half = lane >> 4;
  const int m    = lane & 15;
  const int rowBase = blockIdx.x * 256;
  const int colBase = blockIdx.y * 64;

  v8f acc[2][4] = {};

  // A staging: one 32-wide K row per thread (4x uint4)
  const int agr = rowBase + tid;
  // B staging: thread -> (k-pair, 4-column group)
  const int kp = tid >> 4;          // 0..15  (keys 2kp, 2kp+1)
  const int cg = (tid & 15) << 2;   // 0..60

  for (int k0 = 0; k0 < K; k0 += 32) {
    // --- stage A tile [256 x 32] ---
    {
      U4H8 u[4];
      if (agr < M) {
        const f16* ap = A + (size_t)agr * lda + k0;
        #pragma unroll
        for (int i = 0; i < 4; ++i) u[i].u = *(const uint4*)(ap + 8 * i);
      } else {
        #pragma unroll
        for (int i = 0; i < 4; ++i) u[i].u = make_uint4(0, 0, 0, 0);
      }
      #pragma unroll
      for (int i = 0; i < 4; ++i) *(uint4*)&As[tid * 32 + 8 * i] = u[i].u;
    }
    // --- stage B tile [32 x 64] transposed: packed b32 stores ---
    {
      const f16* bp0 = Bm + (size_t)(k0 + 2 * kp) * ldb + colBase + cg;
      const f16* bp1 = bp0 + ldb;
      U2H4 r0, r1;
      r0.u = *(const uint2*)bp0;
      r1.u = *(const uint2*)bp1;
      #pragma unroll
      for (int e = 0; e < 4; ++e) {
        H2U p; p.h[0] = r0.h[e]; p.h[1] = r1.h[e];
        *(uint*)&Bs[(cg + e) * 32 + 2 * kp] = p.u;
      }
    }
    // Prefetch next K-step tiles (global_prefetch_b8)
    if (k0 + 32 < K) {
      if (agr < M) __builtin_prefetch(A + (size_t)agr * lda + k0 + 32, 0, 0);
      __builtin_prefetch(Bm + (size_t)(k0 + 32 + 2 * kp) * ldb + colBase + cg, 0, 0);
    }
    __syncthreads();

    v16h af[2];
    #pragma unroll
    for (int r = 0; r < 2; ++r) {
      const f16* ar = &As[(32 * wv + 16 * r + m) * 32 + 8 * half];
      af[r] = frag_from_chunks(ar, ar + 16);
    }
    #pragma unroll
    for (int t = 0; t < 4; ++t) {
      const f16* br = &Bs[(16 * t + m) * 32 + 8 * half];
      v16h bf = frag_from_chunks(br, br + 16);
      acc[0][t] = __builtin_amdgcn_wmma_f32_16x16x32_f16(
          false, af[0], false, bf, (short)0, acc[0][t], false, false);
      acc[1][t] = __builtin_amdgcn_wmma_f32_16x16x32_f16(
          false, af[1], false, bf, (short)0, acc[1][t], false, false);
    }
    __syncthreads();
  }

  #pragma unroll
  for (int r = 0; r < 2; ++r) {
    #pragma unroll
    for (int t = 0; t < 4; ++t) {
      #pragma unroll
      for (int j = 0; j < 8; ++j) {
        int gr = rowBase + 32 * wv + 16 * r + j + 8 * half;
        int gc = colBase + 16 * t + m;
        if (gr < M) {
          float v = acc[r][t][j];
          if (bias) v += bias[gc];
          if (ACT == 1) v = 0.5f * v * (1.0f + erff(v * 0.70710678118f));
          if (OUT16) {
            C16[(size_t)gr * ldc + gc] = (f16)v;
          } else if (ADDC) {
            C32[(size_t)gr * ldc + gc] += v;
          } else {
            C32[(size_t)gr * ldc + gc] = v;
          }
        }
      }
    }
  }
}

// ---------------------------------------------------------------------------
// Assemble x[b,t,d]: cls token row, embedded src + emb_b, + positional emb
// ---------------------------------------------------------------------------
__global__ void assemble_kernel(const float* __restrict__ tmp,
                                const float* __restrict__ emb_b,
                                const float* __restrict__ cls,
                                float* __restrict__ x)
{
  size_t idx = (size_t)blockIdx.x * 256 + threadIdx.x;   // B*T*D = 4194304
  int d = (int)(idx & 511);
  int t = (int)((idx >> 9) & 511);
  int b = (int)(idx >> 18);
  float val;
  if (t == 0) val = cls[d];
  else        val = tmp[((size_t)b * SS + (t - 1)) * DD + d] + emb_b[d];
  float je  = (float)(d & ~1);
  float ang = (float)t * __expf(-je * (9.210340371976184f / 512.0f)); // /10000^(je/D)
  val += (d & 1) ? __cosf(ang) : __sinf(ang);
  x[idx] = val;
}

// ---------------------------------------------------------------------------
// LayerNorm row kernel -> f16 output
// ---------------------------------------------------------------------------
__global__ __launch_bounds__(256)
void ln_kernel(const float* __restrict__ x, const float* __restrict__ g,
               const float* __restrict__ bt, f16* __restrict__ out)
{
  int row = blockIdx.x;
  int tid = threadIdx.x;
  const float* xr = x + (size_t)row * DD;
  float a = xr[tid], c = xr[tid + 256];
  __shared__ float r1[256], r2[256];
  r1[tid] = a + c;
  r2[tid] = a * a + c * c;
  __syncthreads();
  for (int s = 128; s > 0; s >>= 1) {
    if (tid < s) { r1[tid] += r1[tid + s]; r2[tid] += r2[tid + s]; }
    __syncthreads();
  }
  float mean = r1[0] * (1.0f / 512.0f);
  float var  = r2[0] * (1.0f / 512.0f) - mean * mean;
  float rstd = rsqrtf(var + 1e-5f);
  out[(size_t)row * DD + tid]       = (f16)((a - mean) * rstd * g[tid]       + bt[tid]);
  out[(size_t)row * DD + tid + 256] = (f16)((c - mean) * rstd * g[tid + 256] + bt[tid + 256]);
}

// ---------------------------------------------------------------------------
// Flash-style attention: one wave per 16-query tile per (b,h).
// Streams 32 keys/iter: 4 WMMA for S = Q*K^T, online softmax, 4 WMMA for
// O += P*V. Adds O/l directly into residual stream x.
// ---------------------------------------------------------------------------
__global__ __launch_bounds__(128)
void attn_kernel(const f16* __restrict__ qf, const f16* __restrict__ kf,
                 const f16* __restrict__ vf, float* __restrict__ x)
{
  const int wv   = threadIdx.x >> 5;
  const int lane = threadIdx.x & 31;
  const int half = lane >> 4;
  const int m    = lane & 15;
  const int qt = blockIdx.x * 4 + wv;
  const int h  = blockIdx.y;
  const int b  = blockIdx.z;

  __shared__ __align__(16) f16 Pb[4][16 * 32];
  __shared__ __align__(16) f16 Vt[4][64 * 32];
  f16* P  = Pb[wv];
  f16* Vs = Vt[wv];

  const size_t hOff = (size_t)h * DHD;

  // Q fragments (rows qt*16 + m), contraction e split 0..31 / 32..63
  v16h qa0, qa1;
  {
    const f16* qp = qf + ((size_t)(b * TT) + qt * 16 + m) * DD + hOff + 8 * half;
    qa0 = frag_from_chunks(qp, qp + 16);
    qa1 = frag_from_chunks(qp + 32, qp + 48);
  }

  v8f o0 = {}, o1 = {}, o2 = {}, o3 = {};
  float mrow[8], lrow[8];
  #pragma unroll
  for (int j = 0; j < 8; ++j) { mrow[j] = -1e30f; lrow[j] = 0.0f; }

  // V staging assignment: key-pair + 32-column group per lane
  const int vkp = lane & 15;          // keys 2*vkp, 2*vkp+1
  const int vcg = (lane >> 4) * 32;   // columns vcg..vcg+31

  for (int kt = 0; kt < TT; kt += 32) {
    // K^T fragments: lane's col = key (kt+m / kt+16+m), contraction = e
    const f16* kp0 = kf + ((size_t)(b * TT) + kt + m) * DD + hOff + 8 * half;
    const f16* kp1 = kp0 + (size_t)16 * DD;
    v16h kb0a = frag_from_chunks(kp0,      kp0 + 16);
    v16h kb0b = frag_from_chunks(kp0 + 32, kp0 + 48);
    v16h kb1a = frag_from_chunks(kp1,      kp1 + 16);
    v16h kb1b = frag_from_chunks(kp1 + 32, kp1 + 48);

    // Prefetch next key/value rows
    if (kt + 32 < TT) {
      __builtin_prefetch(kp0 + (size_t)32 * DD, 0, 0);
      __builtin_prefetch(vf + ((size_t)(b * TT) + kt + 32 + 2 * vkp) * DD + hOff + vcg, 0, 0);
    }

    v8f s0 = {}, s1 = {};
    s0 = __builtin_amdgcn_wmma_f32_16x16x32_f16(false, qa0, false, kb0a, (short)0, s0, false, false);
    s0 = __builtin_amdgcn_wmma_f32_16x16x32_f16(false, qa1, false, kb0b, (short)0, s0, false, false);
    s1 = __builtin_amdgcn_wmma_f32_16x16x32_f16(false, qa0, false, kb1a, (short)0, s1, false, false);
    s1 = __builtin_amdgcn_wmma_f32_16x16x32_f16(false, qa1, false, kb1b, (short)0, s1, false, false);

    // Online softmax per row (row = j + 8*half, data across 16 lanes of half)
    #pragma unroll
    for (int j = 0; j < 8; ++j) {
      float v0 = s0[j] * 0.125f;
      float v1 = s1[j] * 0.125f;
      float mx = fmaxf(v0, v1);
      #pragma unroll
      for (int off = 1; off < 16; off <<= 1)
        mx = fmaxf(mx, __shfl_xor(mx, off, 32));
      float mn = fmaxf(mrow[j], mx);
      float alpha = __expf(mrow[j] - mn);
      mrow[j] = mn;
      float p0 = __expf(v0 - mn);
      float p1 = __expf(v1 - mn);
      float ps = p0 + p1;
      #pragma unroll
      for (int off = 1; off < 16; off <<= 1)
        ps += __shfl_xor(ps, off, 32);
      lrow[j] = lrow[j] * alpha + ps;
      o0[j] *= alpha; o1[j] *= alpha; o2[j] *= alpha; o3[j] *= alpha;
      P[(j + 8*half) * 32 + m]      = (f16)p0;
      P[(j + 8*half) * 32 + 16 + m] = (f16)p1;
    }

    // Stage V tile [32 keys x 64 e] transposed to [e][key], packed b32 stores
    {
      const f16* v0p = vf + ((size_t)(b * TT) + kt + 2 * vkp) * DD + hOff + vcg;
      const f16* v1p = v0p + DD;
      #pragma unroll
      for (int g = 0; g < 4; ++g) {
        U4H8 a, c;
        a.u = *(const uint4*)(v0p + 8 * g);
        c.u = *(const uint4*)(v1p + 8 * g);
        #pragma unroll
        for (int e = 0; e < 8; ++e) {
          H2U p; p.h[0] = a.h[e]; p.h[1] = c.h[e];
          *(uint*)&Vs[(vcg + 8 * g + e) * 32 + 2 * vkp] = p.u;
        }
      }
    }
    __syncthreads();

    // P as A fragment, V^T slices as B fragments; O += P * V
    const f16* pp = &P[m * 32 + 8 * half];
    v16h pa = frag_from_chunks(pp, pp + 16);
    const f16* vr0 = &Vs[(m)      * 32 + 8 * half];
    const f16* vr1 = &Vs[(16 + m) * 32 + 8 * half];
    const f16* vr2 = &Vs[(32 + m) * 32 + 8 * half];
    const f16* vr3 = &Vs[(48 + m) * 32 + 8 * half];
    v16h vb0 = frag_from_chunks(vr0, vr0 + 16);
    v16h vb1 = frag_from_chunks(vr1, vr1 + 16);
    v16h vb2 = frag_from_chunks(vr2, vr2 + 16);
    v16h vb3 = frag_from_chunks(vr3, vr3 + 16);
    o0 = __builtin_amdgcn_wmma_f32_16x16x32_f16(false, pa, false, vb0, (short)0, o0, false, false);
    o1 = __builtin_amdgcn_wmma_f32_16x16x32_f16(false, pa, false, vb1, (short)0, o1, false, false);
    o2 = __builtin_amdgcn_wmma_f32_16x16x32_f16(false, pa, false, vb2, (short)0, o2, false, false);
    o3 = __builtin_amdgcn_wmma_f32_16x16x32_f16(false, pa, false, vb3, (short)0, o3, false, false);
    __syncthreads();
  }

  // O / l added into residual x (each element has exactly one writer)
  #pragma unroll
  for (int j = 0; j < 8; ++j) {
    float inv = 1.0f / lrow[j];
    size_t row = (size_t)(b * TT) + qt * 16 + j + 8 * half;
    float* xp = x + row * DD + hOff;
    xp[m]      += o0[j] * inv;
    xp[16 + m] += o1[j] * inv;
    xp[32 + m] += o2[j] * inv;
    xp[48 + m] += o3[j] * inv;
  }
}

// ---------------------------------------------------------------------------
// Head: sigmoid(x[:,0] @ head_W + head_b), one block per batch row
// ---------------------------------------------------------------------------
__global__ __launch_bounds__(256)
void head_kernel(const float* __restrict__ x, const float* __restrict__ hw,
                 const float* __restrict__ hb, float* __restrict__ out)
{
  int b = blockIdx.x, tid = threadIdx.x;
  __shared__ float red[256];
  const float* xr = x + (size_t)b * TT * DD;
  red[tid] = xr[tid] * hw[tid] + xr[tid + 256] * hw[tid + 256];
  __syncthreads();
  for (int s = 128; s > 0; s >>= 1) {
    if (tid < s) red[tid] += red[tid + s];
    __syncthreads();
  }
  if (tid == 0) out[b] = 1.0f / (1.0f + __expf(-(red[0] + hb[0])));
}

// ---------------------------------------------------------------------------
extern "C" void kernel_launch(void* const* d_in, const int* in_sizes, int n_in,
                              void* d_out, int out_size, void* d_ws, size_t ws_size,
                              hipStream_t stream)
{
  const float* src   = (const float*)d_in[0];
  const float* emb_W = (const float*)d_in[1];
  const float* emb_b = (const float*)d_in[2];
  const float* cls   = (const float*)d_in[3];
  const float* ln1_g = (const float*)d_in[4];
  const float* ln1_b = (const float*)d_in[5];
  const float* Wq    = (const float*)d_in[6];
  const float* bq    = (const float*)d_in[7];
  const float* Wk    = (const float*)d_in[8];
  const float* bk    = (const float*)d_in[9];
  const float* Wv    = (const float*)d_in[10];
  const float* bv    = (const float*)d_in[11];
  const float* ln2_g = (const float*)d_in[12];
  const float* ln2_b = (const float*)d_in[13];
  const float* W1    = (const float*)d_in[14];
  const float* b1    = (const float*)d_in[15];
  const float* W2    = (const float*)d_in[16];
  const float* b2    = (const float*)d_in[17];
  const float* headW = (const float*)d_in[18];
  const float* headb = (const float*)d_in[19];
  float* out = (float*)d_out;

  const size_t ROWS = (size_t)BB * TT;      // 8192
  char* w = (char*)d_ws;
  float* xbuf  = (float*)w;  w += ROWS * DD * 4;            // 16 MB
  f16*   hf    = (f16*)w;    w += ROWS * DD * 2;            // 8 MB
  f16*   qfb   = (f16*)w;    w += ROWS * DD * 2;            // 8 MB
  f16*   kfb   = (f16*)w;    w += ROWS * DD * 2;            // 8 MB
  f16*   vfb   = (f16*)w;    w += ROWS * DD * 2;            // 8 MB
  f16*   hidden= (f16*)w;    w += ROWS * MLPD * 2;          // 32 MB
  f16*   wEmb  = (f16*)w;    w += (size_t)FFT * DD * 2;
  f16*   wQf   = (f16*)w;    w += (size_t)NBLK * HH * DHD * DHD * 2;
  f16*   wKf   = (f16*)w;    w += (size_t)NBLK * HH * DHD * DHD * 2;
  f16*   wVf   = (f16*)w;    w += (size_t)NBLK * HH * DHD * DHD * 2;
  f16*   w1f   = (f16*)w;    w += (size_t)NBLK * DD * MLPD * 2;
  f16*   w2f   = (f16*)w;    w += (size_t)NBLK * DD * MLPD * 2;
  float* tmpEmb = (float*)hidden;  // alias (disjoint lifetime)
  f16*   srcf   = qfb;             // alias (disjoint lifetime)

  auto cvt = [&](const float* in, f16* o, long n) {
    int grid = (int)((n + 1023) / 1024);
    if (grid > 8192) grid = 8192;
    f32_to_f16<<<grid, 256, 0, stream>>>(in, o, n);
  };
  cvt(src,   srcf, (long)BB * SS * FFT);
  cvt(emb_W, wEmb, (long)FFT * DD);
  cvt(Wq,    wQf,  (long)NBLK * HH * DHD * DHD);
  cvt(Wk,    wKf,  (long)NBLK * HH * DHD * DHD);
  cvt(Wv,    wVf,  (long)NBLK * HH * DHD * DHD);
  cvt(W1,    w1f,  (long)NBLK * DD * MLPD);
  cvt(W2,    w2f,  (long)NBLK * DD * MLPD);

  // Embedding GEMM: [8176,128] @ [128,512] -> tmpEmb
  wmma_gemm<0, false, false><<<dim3(32, 8, 1), 256, 0, stream>>>(
      srcf, FFT, wEmb, DD, tmpEmb, nullptr, DD, nullptr,
      BB * SS, DD, FFT, 0, 0, 0, 0);

  // Assemble x with cls token, emb bias, positional embeddings
  assemble_kernel<<<16384, 256, 0, stream>>>(tmpEmb, emb_b, cls, xbuf);

  for (int i = 0; i < NBLK; ++i) {
    // LN1 -> hf (f16)
    ln_kernel<<<(int)ROWS, 256, 0, stream>>>(xbuf, ln1_g + i * DD, ln1_b + i * DD, hf);

    // QKV per head: M=8192, N=64, K=64; z indexes heads
    const int hwS = HH * DHD * DHD;     // per-block weight stride
    wmma_gemm<0, false, true><<<dim3(32, 1, HH), 256, 0, stream>>>(
        hf, DD, wQf + (size_t)i * hwS, DHD, nullptr, qfb, DD, bq + (size_t)i * HH * DHD,
        (int)ROWS, DHD, DHD, DHD, DHD * DHD, DHD, DHD);
    wmma_gemm<0, false, true><<<dim3(32, 1, HH), 256, 0, stream>>>(
        hf, DD, wKf + (size_t)i * hwS, DHD, nullptr, kfb, DD, bk + (size_t)i * HH * DHD,
        (int)ROWS, DHD, DHD, DHD, DHD * DHD, DHD, DHD);
    wmma_gemm<0, false, true><<<dim3(32, 1, HH), 256, 0, stream>>>(
        hf, DD, wVf + (size_t)i * hwS, DHD, nullptr, vfb, DD, bv + (size_t)i * HH * DHD,
        (int)ROWS, DHD, DHD, DHD, DHD * DHD, DHD, DHD);

    // Flash attention, adds into xbuf (residual)
    attn_kernel<<<dim3(TT / 64, HH, BB), 128, 0, stream>>>(qfb, kfb, vfb, xbuf);

    // LN2 -> hf (f16)
    ln_kernel<<<(int)ROWS, 256, 0, stream>>>(xbuf, ln2_g + i * DD, ln2_b + i * DD, hf);

    // MLP1: [8192,512]@[512,2048] + b1, exact GELU, f16 out
    wmma_gemm<1, false, true><<<dim3(32, MLPD / 64, 1), 256, 0, stream>>>(
        hf, DD, w1f + (size_t)i * DD * MLPD, MLPD, nullptr, hidden, MLPD,
        b1 + (size_t)i * MLPD, (int)ROWS, MLPD, DD, 0, 0, 0, 0);

    // MLP2: [8192,2048]@[2048,512] + b2, accumulate into xbuf (residual)
    wmma_gemm<0, true, false><<<dim3(32, DD / 64, 1), 256, 0, stream>>>(
        hidden, MLPD, w2f + (size_t)i * DD * MLPD, DD, xbuf, nullptr, DD,
        b2 + (size_t)i * DD, (int)ROWS, DD, MLPD, 0, 0, 0, 0);
  }

  head_kernel<<<BB, 256, 0, stream>>>(xbuf, headW, headb, out);
}
